// GINEEncoder_19628000542880
// MI455X (gfx1250) — matrix operand order repaired and checked by reading, not compiled, not run on hardware
//
#include <hip/hip_runtime.h>
#include <hip/hip_bf16.h>

typedef __attribute__((ext_vector_type(16))) _Float16 v16h;
typedef __attribute__((ext_vector_type(8)))  float    v8f;

#define N_NODES  100000
#define N_EDGES  640000
#define N_GRAPHS 512
#define NODE_IN  64
#define EDGE_IN  16
#define HID      128
#define NLAYERS  4
#define BN_EPS   1e-5f

// ---------------------------------------------------------------------------
// Weight packing: W [Ksrc x N] f32 row-major -> per-lane WMMA B-operand tiles,
// grouped so the 4 n-tiles consumed together by one wave are contiguous.
// Lane L of tile (nt, kt) holds 16 halfs:
//   b[j] = W[kt*32 + (L>>4)*8 + (j&7) + (j>>3)*16][nt*16 + (L&15)]   (zero-pad k >= Ksrc)
// Linear layout: P[(((ntg*nkt + kt)*4 + t)*32 + lane)*16 + j],
//   where nt = ntg*4 + t.  Requires N % 64 == 0.
// ---------------------------------------------------------------------------
__global__ __launch_bounds__(256) void pack_b_kernel(const float* __restrict__ W,
                                                     _Float16* __restrict__ P,
                                                     int Ksrc, int Kp, int N) {
  int tid = blockIdx.x * blockDim.x + threadIdx.x;
  int total = Kp * N;
  if (tid >= total) return;
  int nkt  = Kp >> 5;
  int tile = tid >> 9;          // 512 halfs per 32x16 tile
  int r    = tid & 511;
  int lane = r >> 4;
  int j    = r & 15;
  int t    = tile & 3;          // n-tile within group of 4
  int g    = tile >> 2;
  int ntg  = g / nkt;
  int kt   = g % nkt;
  int nt   = ntg * 4 + t;
  int k = kt * 32 + (lane >> 4) * 8 + (j & 7) + (j >> 3) * 16;
  int n = nt * 16 + (lane & 15);
  float v = (k < Ksrc) ? W[(size_t)k * N + n] : 0.0f;
  P[tid] = (_Float16)v;
}

// ---------------------------------------------------------------------------
// WMMA GEMM: one wave computes a 16x64 strip (4 n-tiles) of D = A*B + bias.
// A-modes:  0: A f32 row-major            1: A = A0 + A1 (both f32, fused add)
//           2: A f16 row-major            3: A f32 with K=16 zero-padded to 32
// Out-modes:0: relu -> f32                1: relu -> f16        2: plain -> f32
// A per-lane layout (16x32 f16): m = lane&15, hs = lane>>4,
//   a[i] = A[m][k0+hs*8+i], a[8+i] = A[m][k0+16+hs*8+i]  (i=0..7)
// C/D per-lane layout (16x16 f32): col n = lane&15, acc[r] = D[hs*8+r][n]
// One A fragment feeds 4 back-to-back v_wmma_f32_16x16x32_f16.
// ---------------------------------------------------------------------------
template <int AM, int OM>
__global__ __launch_bounds__(256) void gemm_wmma_kernel(
    const float* __restrict__ A0, const float* __restrict__ A1,
    const _Float16* __restrict__ Ah, const _Float16* __restrict__ Bp,
    const float* __restrict__ bias, float* __restrict__ Of,
    _Float16* __restrict__ Oh, int Mt, int N, int K, int lda, int ldc) {
  int wid  = (blockIdx.x * 256 + (int)threadIdx.x) >> 5;  // wave-uniform
  int lane = threadIdx.x & 31;
  int ng   = N >> 6;            // groups of 4 n-tiles
  int mt   = wid / ng;
  int ntg  = wid % ng;
  if (mt >= Mt) return;  // whole-wave exit: EXEC stays all-ones for WMMA

  int hs  = lane >> 4;
  int l15 = lane & 15;
  int m   = mt * 16 + l15;      // A row for this lane
  int nkt = K >> 5;
  int nt0 = ntg * 4;

  v8f acc[4];
#pragma unroll
  for (int t = 0; t < 4; ++t) {
    float bv = bias[(nt0 + t) * 16 + l15];
#pragma unroll
    for (int r = 0; r < 8; ++r) acc[t][r] = bv;
  }

  const v16h* Bv = (const v16h*)Bp + ((size_t)ntg * nkt * 4) * 32 + lane;

  for (int kt = 0; kt < nkt; ++kt) {
    int k0 = kt * 32 + hs * 8;
    v16h a;
    if (AM == 0) {
      const float* p = A0 + (size_t)m * lda + k0;
#pragma unroll
      for (int i = 0; i < 8; ++i) {
        a[i]     = (_Float16)p[i];
        a[8 + i] = (_Float16)p[16 + i];
      }
    } else if (AM == 1) {
      const float* p = A0 + (size_t)m * lda + k0;
      const float* q = A1 + (size_t)m * lda + k0;
#pragma unroll
      for (int i = 0; i < 8; ++i) {
        a[i]     = (_Float16)(p[i] + q[i]);
        a[8 + i] = (_Float16)(p[16 + i] + q[16 + i]);
      }
    } else if (AM == 2) {
      const _Float16* p = Ah + (size_t)m * lda + k0;
#pragma unroll
      for (int i = 0; i < 8; ++i) {
        a[i]     = p[i];
        a[8 + i] = p[16 + i];
      }
    } else {  // AM == 3: K=16 source zero-padded to K=32
      const float* p = A0 + (size_t)m * lda + hs * 8;
#pragma unroll
      for (int i = 0; i < 8; ++i) {
        a[i]     = (_Float16)p[i];
        a[8 + i] = (_Float16)0.0f;
      }
    }
    v16h b0 = Bv[0 * 32];
    v16h b1 = Bv[1 * 32];
    v16h b2 = Bv[2 * 32];
    v16h b3 = Bv[3 * 32];
    // speculative prefetch of next k-step's B group (silently dropped if OOB)
    __builtin_prefetch(Bv + 4 * 32, 0, 0);
    acc[0] = __builtin_amdgcn_wmma_f32_16x16x32_f16(false, a, false, b0,
                                                    (short)0, acc[0], false, false);
    acc[1] = __builtin_amdgcn_wmma_f32_16x16x32_f16(false, a, false, b1,
                                                    (short)0, acc[1], false, false);
    acc[2] = __builtin_amdgcn_wmma_f32_16x16x32_f16(false, a, false, b2,
                                                    (short)0, acc[2], false, false);
    acc[3] = __builtin_amdgcn_wmma_f32_16x16x32_f16(false, a, false, b3,
                                                    (short)0, acc[3], false, false);
    Bv += 4 * 32;
  }

  int mrow = mt * 16 + hs * 8;
#pragma unroll
  for (int t = 0; t < 4; ++t) {
    int n = (nt0 + t) * 16 + l15;
#pragma unroll
    for (int r = 0; r < 8; ++r) {
      float v = acc[t][r];
      if (OM == 0 || OM == 1) v = fmaxf(v, 0.0f);
      size_t idx = (size_t)(mrow + r) * ldc + n;
      if (OM == 1) Oh[idx] = (_Float16)v;
      else         Of[idx] = v;
    }
  }
}

// ---------------------------------------------------------------------------
__global__ __launch_bounds__(256) void fill_zero_kernel(float* __restrict__ p, int n) {
  int i = blockIdx.x * blockDim.x + threadIdx.x;
  if (i < n) p[i] = 0.0f;
}

// msg = relu(h[src] + e); atomic scatter-add into agg[dst]. 32 threads/edge.
__global__ __launch_bounds__(256) void msg_scatter_kernel(
    const float* __restrict__ h, const _Float16* __restrict__ e,
    const int* __restrict__ src, const int* __restrict__ dst,
    float* __restrict__ agg, int E) {
  int tid = blockIdx.x * blockDim.x + threadIdx.x;
  int eid = tid >> 5;
  if (eid >= E) return;
  int c4 = (tid & 31) * 4;
  int s = src[eid], d = dst[eid];
  const float*    hp = h   + (size_t)s   * HID + c4;
  const _Float16* ep = e   + (size_t)eid * HID + c4;
  float*          ap = agg + (size_t)d   * HID + c4;
#pragma unroll
  for (int i = 0; i < 4; ++i) {
    float v = hp[i] + (float)ep[i];
    v = fmaxf(v, 0.0f);
    unsafeAtomicAdd(ap + i, v);   // global_atomic_add_f32
  }
}

// Per-column sum and sum-of-squares over M rows -> stats[0:128]=sum, [128:256]=sumsq
__global__ __launch_bounds__(256) void bn_stats_kernel(const float* __restrict__ z,
                                                       float* __restrict__ stats, int M) {
  __shared__ float ls[256];
  __shared__ float lq[256];
  int c    = threadIdx.x & 127;
  int half = threadIdx.x >> 7;
  int walker = blockIdx.x * 2 + half;
  int stride = gridDim.x * 2;
  float s = 0.0f, q = 0.0f;
  for (int r = walker; r < M; r += stride) {
    float v = z[(size_t)r * HID + c];
    s += v;
    q += v * v;
  }
  ls[threadIdx.x] = s;
  lq[threadIdx.x] = q;
  __syncthreads();
  if (half == 0) {
    s = ls[threadIdx.x] + ls[threadIdx.x + 128];
    q = lq[threadIdx.x] + lq[threadIdx.x + 128];
    unsafeAtomicAdd(&stats[c], s);
    unsafeAtomicAdd(&stats[128 + c], q);
  }
}

// h = relu((z - mean) * rsqrt(var + eps) * gamma + beta), biased var
__global__ __launch_bounds__(256) void bn_apply_kernel(
    const float* __restrict__ z, const float* __restrict__ stats,
    const float* __restrict__ gamma, const float* __restrict__ beta,
    float* __restrict__ h, int M) {
  int i = blockIdx.x * blockDim.x + threadIdx.x;
  if (i >= M * HID) return;
  int c = i & 127;
  float inv_n = 1.0f / (float)M;
  float mean = stats[c] * inv_n;
  float var  = stats[128 + c] * inv_n - mean * mean;
  float isd  = rsqrtf(var + BN_EPS);
  float v = (z[i] - mean) * isd * gamma[c] + beta[c];
  h[i] = fmaxf(v, 0.0f);
}

// Segment-sum h by graph id + per-graph counts. 32 threads/node.
__global__ __launch_bounds__(256) void pool_scatter_kernel(
    const float* __restrict__ h, const int* __restrict__ batch,
    float* __restrict__ sums, float* __restrict__ cnt, int M) {
  int tid = blockIdx.x * blockDim.x + threadIdx.x;
  int node = tid >> 5;
  if (node >= M) return;
  int c4 = (tid & 31) * 4;
  int g = batch[node];
  const float* hp = h    + (size_t)node * HID + c4;
  float*       sp = sums + (size_t)g    * HID + c4;
#pragma unroll
  for (int i = 0; i < 4; ++i) unsafeAtomicAdd(sp + i, hp[i]);
  if ((tid & 31) == 0) unsafeAtomicAdd(cnt + g, 1.0f);
}

__global__ __launch_bounds__(256) void pool_final_kernel(
    const float* __restrict__ sums, const float* __restrict__ cnt,
    float* __restrict__ out) {
  int i = blockIdx.x * blockDim.x + threadIdx.x;
  if (i >= N_GRAPHS * HID) return;
  int g = i >> 7;
  out[i] = sums[i] / fmaxf(cnt[g], 1.0f);
}

// ---------------------------------------------------------------------------
extern "C" void kernel_launch(void* const* d_in, const int* in_sizes, int n_in,
                              void* d_out, int out_size, void* d_ws, size_t ws_size,
                              hipStream_t stream) {
  (void)in_sizes; (void)n_in; (void)out_size; (void)ws_size;
  const float* x         = (const float*)d_in[0];
  const int*   edge_idx  = (const int*)  d_in[1];
  const float* edge_attr = (const float*)d_in[2];
  const int*   batch     = (const int*)  d_in[3];
  const float* node_w    = (const float*)d_in[4];
  const float* node_b    = (const float*)d_in[5];
  const float* edge_w    = (const float*)d_in[6];
  const float* edge_b    = (const float*)d_in[7];
  const float* w1        = (const float*)d_in[8];
  const float* b1        = (const float*)d_in[9];
  const float* w2        = (const float*)d_in[10];
  const float* b2        = (const float*)d_in[11];
  const float* gamma     = (const float*)d_in[12];
  const float* beta      = (const float*)d_in[13];
  float* out = (float*)d_out;

  const int* src = edge_idx;
  const int* dst = edge_idx + N_EDGES;

  // Workspace carve-up (256B aligned)
  char* base = (char*)d_ws;
  size_t off = 0;
  auto carve = [&](size_t bytes) -> char* {
    off = (off + 255) & ~(size_t)255;
    char* p = base + off;
    off += bytes;
    return p;
  };
  float*    h       = (float*)   carve((size_t)N_NODES * HID * 4);       // node features
  float*    agg     = (float*)   carve((size_t)N_NODES * HID * 4);       // scatter accum
  float*    z2      = (float*)   carve((size_t)N_NODES * HID * 4);       // pre-BN
  _Float16* e_f16   = (_Float16*)carve((size_t)N_EDGES * HID * 2);       // edge features
  _Float16* t_f16   = (_Float16*)carve((size_t)N_NODES * 2 * HID * 2);   // MLP hidden
  _Float16* pw_node = (_Float16*)carve((size_t)64  * 128 * 2);
  _Float16* pw_edge = (_Float16*)carve((size_t)32  * 128 * 2);
  _Float16* pw_w1   = (_Float16*)carve((size_t)NLAYERS * 128 * 256 * 2);
  _Float16* pw_w2   = (_Float16*)carve((size_t)NLAYERS * 256 * 128 * 2);
  float*    stats   = (float*)   carve(256 * 4);
  float*    psums   = (float*)   carve((size_t)N_GRAPHS * HID * 4);
  float*    pcnt    = (float*)   carve((size_t)N_GRAPHS * 4);

  auto cdiv = [](long long a, long long b) { return (int)((a + b - 1) / b); };

  // ---- pack all weights into WMMA B-operand layout (f16) ----
  pack_b_kernel<<<cdiv(64 * 128, 256), 256, 0, stream>>>(node_w, pw_node, 64, 64, 128);
  pack_b_kernel<<<cdiv(32 * 128, 256), 256, 0, stream>>>(edge_w, pw_edge, 16, 32, 128);
  for (int l = 0; l < NLAYERS; ++l) {
    pack_b_kernel<<<cdiv(128 * 256, 256), 256, 0, stream>>>(
        w1 + (size_t)l * 128 * 256, pw_w1 + (size_t)l * 128 * 256, 128, 128, 256);
    pack_b_kernel<<<cdiv(256 * 128, 256), 256, 0, stream>>>(
        w2 + (size_t)l * 256 * 128, pw_w2 + (size_t)l * 256 * 128, 256, 256, 128);
  }

  // ---- input projections ----
  {   // h = relu(x @ node_w + node_b)  : M=100000 K=64 N=128
    int waves = (N_NODES / 16) * (HID / 64);
    gemm_wmma_kernel<0, 0><<<cdiv(waves, 8), 256, 0, stream>>>(
        x, nullptr, nullptr, pw_node, node_b, h, nullptr,
        N_NODES / 16, HID, 64, NODE_IN, HID);
  }
  {   // e = relu(edge_attr @ edge_w + edge_b) : M=640000 K=16(pad 32) N=128, f16 out
    int waves = (N_EDGES / 16) * (HID / 64);
    gemm_wmma_kernel<3, 1><<<cdiv(waves, 8), 256, 0, stream>>>(
        edge_attr, nullptr, nullptr, pw_edge, edge_b, nullptr, e_f16,
        N_EDGES / 16, HID, 32, EDGE_IN, HID);
  }

  // ---- GINE layers ----
  for (int l = 0; l < NLAYERS; ++l) {
    fill_zero_kernel<<<cdiv((long long)N_NODES * HID, 256), 256, 0, stream>>>(
        agg, N_NODES * HID);
    msg_scatter_kernel<<<cdiv((long long)N_EDGES * 32, 256), 256, 0, stream>>>(
        h, e_f16, src, dst, agg, N_EDGES);
    {   // t = relu((h + agg) @ w1[l] + b1[l]) : K=128 N=256, f16 out
      int waves = (N_NODES / 16) * (256 / 64);
      gemm_wmma_kernel<1, 1><<<cdiv(waves, 8), 256, 0, stream>>>(
          h, agg, nullptr, pw_w1 + (size_t)l * 128 * 256, b1 + (size_t)l * 256,
          nullptr, t_f16, N_NODES / 16, 256, 128, HID, 256);
    }
    {   // z2 = t @ w2[l] + b2[l] : K=256 N=128, f32 out (no relu)
      int waves = (N_NODES / 16) * (HID / 64);
      gemm_wmma_kernel<2, 2><<<cdiv(waves, 8), 256, 0, stream>>>(
          nullptr, nullptr, t_f16, pw_w2 + (size_t)l * 256 * 128,
          b2 + (size_t)l * HID, z2, nullptr, N_NODES / 16, HID, 256, 256, HID);
    }
    fill_zero_kernel<<<1, 256, 0, stream>>>(stats, 256);
    bn_stats_kernel<<<512, 256, 0, stream>>>(z2, stats, N_NODES);
    bn_apply_kernel<<<cdiv((long long)N_NODES * HID, 256), 256, 0, stream>>>(
        z2, stats, gamma + (size_t)l * HID, beta + (size_t)l * HID, h, N_NODES);
  }

  // ---- global mean pool ----
  fill_zero_kernel<<<cdiv(N_GRAPHS * HID, 256), 256, 0, stream>>>(psums, N_GRAPHS * HID);
  fill_zero_kernel<<<1, 512, 0, stream>>>(pcnt, N_GRAPHS);
  pool_scatter_kernel<<<cdiv((long long)N_NODES * 32, 256), 256, 0, stream>>>(
      h, batch, psums, pcnt, N_NODES);
  pool_final_kernel<<<cdiv(N_GRAPHS * HID, 256), 256, 0, stream>>>(psums, pcnt, out);
}
